// NewNodeFeaturesFrom3Body_76510547411072
// MI455X (gfx1250) — matrix operand order repaired and checked by reading, not compiled, not run on hardware
//
#include <hip/hip_runtime.h>

// ---------------------------------------------------------------------------
// NewNodeFeaturesFrom3Body on MI455X (gfx1250, wave32, WMMA).
// Pipeline:
//   K0 pack_w:       W0/W1/W2 (48x384 f32) -> bf16 WMMA-B fragments (1/sqrt48 folded)
//   K1 linear_wmma:  scrambled gather of node_attr -> LDS in per-(l,m) u-major
//                    layout (zero-padded K 48->64), WMMA-A fragments via 4x
//                    ds_load_b128, 9x (per (l,m)) GEMMs via v_wmma_f32_16x16x32_bf16,
//                    COALESCED stores in natural layout yflat[n][lm*384+v].
//   K2 cg:           per (n,channel): applies the (N,3456)->(N,384,9) reshape scramble
//                    on the READ side, then 3x sparse real-SH Clebsch-Gordan
//                    contraction (hard-coded coefficients) + w_body/sqrt(3) -> d_out.
// Workspace: [0,147456) bf16 B-fragments; [147456, +N*3456*4) yflat f32 (~276.6 MB).
// ---------------------------------------------------------------------------

typedef __attribute__((ext_vector_type(16))) __bf16 v16bf;
typedef __attribute__((ext_vector_type(8)))  __bf16 v8bf;
typedef __attribute__((ext_vector_type(8)))  float  v8f;

#define ZDIM 144
#define CCH  128
#define DDIM 9
#define ACOL 432          // 3*ZDIM
#define MROW 384          // 3*CCH
#define YROW 3456         // 9*384 per node
#define KPAD 64           // u (=K) padded 48 -> 64
#define ROWSTR (9*KPAD + 8)  // 584 elems; keeps 16B align, breaks bank conflicts
#define WFRAG_ELEMS (3*2*24*32*16)   // l * kstep * vtile * lane * elem = 73728 bf16
#define WFULL_OFF   (WFRAG_ELEMS*2)  // 147456 bytes

// ---------------- K0: pack weights into WMMA B-operand fragments ----------
// B (16-bit, 32x16 K x N) lane layout: lanes 0-15 hold K=0..7,16..23 (elem e -> K=(e&7)+16*(e>>3)),
// lanes 16-31 hold K=8..15,24..31 (elem e -> K=8+(e&7)+16*(e>>3)); +32 for kstep 1; N = vtile*16 + lane%16.
__global__ __launch_bounds__(256) void pack_w_kernel(
    const float* __restrict__ W0, const float* __restrict__ W1,
    const float* __restrict__ W2, unsigned short* __restrict__ wfrag_u)
{
    int idx = blockIdx.x * 256 + threadIdx.x;
    if (idx >= WFRAG_ELEMS) return;
    __bf16* wfrag = (__bf16*)wfrag_u;
    int e    = idx & 15;
    int lane = (idx >> 4) & 31;
    int t    = idx >> 9;            // (l*2+ks)*24 + vt
    int vt   = t % 24;
    int lk   = t / 24;
    int ks   = lk & 1;
    int l    = lk >> 1;
    int K = ks * 32 + ((lane >> 4) << 3) + (e & 7) + ((e >> 3) << 4);
    int v = vt * 16 + (lane & 15);
    float val = 0.0f;
    if (K < 48) {
        const float* W = (l == 0) ? W0 : ((l == 1) ? W1 : W2);
        val = W[K * MROW + v] * 0.14433756729740643f;   // 1/sqrt(48)
    }
    wfrag[idx] = (__bf16)val;
}

// ---------------- K1: gather + WMMA GEMM + coalesced natural-layout store --
__global__ __launch_bounds__(256) void linear_wmma_kernel(
    const float* __restrict__ attr, const unsigned short* __restrict__ wfrag_u,
    float* __restrict__ yflat, int N)
{
    // A-tile, u-major per (row, lm): Alds[row*ROWSTR + lm*64 + u] = nab[n0+row, col(lm,u)]
    __shared__ __attribute__((aligned(16))) __bf16 Alds[16 * ROWSTR];  // 18.7 KB
    const int n0  = blockIdx.x * 16;
    const int tid = threadIdx.x;
    const int ZN  = ZDIM * N;

    // Per-lm block geometry: col = s0 + u*dim + m  (u >= 48 zero-padded).
    const int s0_t[9]  = {0, 48, 48, 48, 192, 192, 192, 192, 192};
    const int dim_t[9] = {1, 3, 3, 3, 5, 5, 5, 5, 5};
    const int m_t[9]   = {0, 0, 1, 2, 0, 1, 2, 3, 4};

    // Gather: nab[r, col] = attr[n', z'] with L = r*432+col, b=L/ZN, rem=L-b*ZN,
    // z'=rem/N, n'=rem-z'*N  (faithful torch (3,Z,N)->(N,3Z) reshape semantics).
    for (int idx = tid; idx < 16 * 9 * KPAD; idx += 256) {
        int row = idx / (9 * KPAD);
        int rem9 = idx - row * (9 * KPAD);
        int lm  = rem9 >> 6;
        int u   = rem9 & 63;
        int nrow = n0 + row;
        __bf16 val = (__bf16)0.0f;
        if (u < 48 && nrow < N) {
            int col = s0_t[lm] + u * dim_t[lm] + m_t[lm];
            int L   = nrow * ACOL + col;
            int b   = L / ZN;
            int rm  = L - b * ZN;
            int z   = rm / N;
            int nn  = rm - z * N;
            val = (__bf16)attr[nn * ZDIM + z];
        }
        Alds[row * ROWSTR + lm * KPAD + u] = val;
    }
    __syncthreads();

    const int wave  = tid >> 5;
    const int lane  = tid & 31;
    const int laneM = lane & 15;
    const int hi    = lane >> 4;
    const bool fulltile = (n0 + 16 <= N);
    const v16bf* bfr = (const v16bf*)wfrag_u;

    for (int lm = 0; lm < 9; ++lm) {
        // A (16-bit, 16x32 M x K) lane layout: M = lane%16; elems 0..7 -> K=hi*8+0..7,
        // elems 8..15 -> K=hi*8+16..23 (+32 for kstep 1; zeros come from LDS padding).
        const __bf16* arow = &Alds[laneM * ROWSTR + lm * KPAD + hi * 8];
        v8bf p0 = *(const v8bf*)(arow);        // K = hi*8 + 0..7
        v8bf p1 = *(const v8bf*)(arow + 16);   // K = hi*8 + 16..23
        v8bf p2 = *(const v8bf*)(arow + 32);   // K = hi*8 + 32..39
        v8bf p3 = *(const v8bf*)(arow + 48);   // K = hi*8 + 48..55 (zeros)
        v16bf a0 = __builtin_shufflevector(p0, p1, 0,1,2,3,4,5,6,7,8,9,10,11,12,13,14,15);
        v16bf a1 = __builtin_shufflevector(p2, p3, 0,1,2,3,4,5,6,7,8,9,10,11,12,13,14,15);

        #pragma unroll
        for (int i = 0; i < 3; ++i) {               // 24 v-tiles, 8 waves -> 3 each
            int vt = wave + i * 8;
            v16bf b0 = bfr[((lm < 1 ? 0 : (lm < 4 ? 1 : 2)) * 2 + 0) * 24 * 32 + vt * 32 + lane];
            v16bf b1 = bfr[((lm < 1 ? 0 : (lm < 4 ? 1 : 2)) * 2 + 1) * 24 * 32 + vt * 32 + lane];
            v8f c = {};
            c = __builtin_amdgcn_wmma_f32_16x16x32_bf16(false, a0, false, b0, (short)0, c, false, false);
            c = __builtin_amdgcn_wmma_f32_16x16x32_bf16(false, a1, false, b1, (short)0, c, false, false);

            // Natural layout: yflat[n][lm*384 + v]; per e, each half-wave writes a
            // contiguous 16-float (64B) segment -> fully coalesced.
            int base = lm * MROW + vt * 16 + laneM;
            if (fulltile) {
                #pragma unroll
                for (int e = 0; e < 8; ++e)
                    yflat[(n0 + e + hi * 8) * YROW + base] = c[e];
            } else {
                #pragma unroll
                for (int e = 0; e < 8; ++e) {
                    int n = n0 + e + hi * 8;
                    if (n < N) yflat[n * YROW + base] = c[e];
                }
            }
        }
    }
}

// ---------------- K2: scramble-on-read + sparse CG + body reduction --------
__global__ __launch_bounds__(256) void cg_kernel(
    const float* __restrict__ nbody, const float* __restrict__ yflat,
    const float* __restrict__ wbody, float* __restrict__ out, int N)
{
    int gid = blockIdx.x * 256 + threadIdx.x;
    if (gid >= N * CCH) return;
    int n = gid / CCH;
    int c = gid - n * CCH;

    const float* xp = nbody + gid * 27;        // x[d*3+t] = node_nbody[n,c,d,t]
    const float* yb = yflat + n * YROW;        // [lm*384 + v]

    float x[27];
    #pragma unroll
    for (int i = 0; i < 27; ++i) x[i] = xp[i];

    // Reshape scramble on the read side: w[t*9+d] for j = 3c+t.
    //  d=0:    y0[v=j]                         -> yb[3c+t]
    //  d=1..3: y1[m=t][v=(d-1)*128+c]          -> yb[(1+t)*384 + (d-1)*128 + c]
    //  d=4..8: q=(d-4)*384+3c+t; v=q/5; m=q%5  -> yb[(4+m)*384 + v]
    float w[27];
    #pragma unroll
    for (int t = 0; t < 3; ++t) {
        w[t * 9 + 0] = yb[3 * c + t];
        #pragma unroll
        for (int d = 1; d <= 3; ++d)
            w[t * 9 + d] = yb[(1 + t) * MROW + (d - 1) * 128 + c];
        #pragma unroll
        for (int d = 4; d <= 8; ++d) {
            int q = (d - 4) * MROW + 3 * c + t;
            int v = q / 5;
            int m = q - 5 * v;
            w[t * 9 + d] = yb[(4 + m) * MROW + v];
        }
    }

    const float IS3 = 0.57735026918962576f;   // 1/sqrt(3)
    float wb0 = wbody[0] * IS3, wb1 = wbody[1] * IS3, wb2 = wbody[2] * IS3;

    // Real-SH CG constants (derived analytically from the reference's _real_cg):
    const float C110 = 0.57735026918962576f;  // (1,1,0): -C110 * delta
    const float C220 = 0.44721359549995794f;  // (2,2,0): +C220 * delta
    const float S  = 0.54772255750516611f;    // sqrt(3/10)
    const float T  = 0.31622776601683794f;    // sqrt(1/10)
    const float U  = 0.63245553203367590f;    // sqrt(2/5)
    const float R  = 0.70710678118654752f;    // 1/sqrt(2)
    const float G  = 0.81649658092772603f;    // 2/sqrt(6)
    const float H  = 0.40824829046386302f;    // 1/sqrt(6)
    const float P  = 0.53452248382484879f;    // lambda*(-2/sqrt3) = sqrt(2/7)
    const float Q  = -0.46291004988627577f;   // lambda = -sqrt(3/14)
    const float RR = -0.26726124191242438f;   // lambda/sqrt3 = -sqrt(1/14)

    float acc[9];
    #pragma unroll
    for (int d = 0; d < 9; ++d) acc[d] = 0.0f;

    #pragma unroll
    for (int t = 0; t < 3; ++t) {
        const float* W = w + t * 9;
        float x0 = x[0+t], x1 = x[3+t], x2 = x[6+t],  x3 = x[9+t],  x4 = x[12+t],
              x5 = x[15+t], x6 = x[18+t], x7 = x[21+t], x8 = x[24+t];
        float w0 = W[0], w1 = W[1], w2 = W[2], w3 = W[3], w4 = W[4],
              w5 = W[5], w6 = W[6], w7 = W[7], w8 = W[8];

        // l3=0 (paths (0,0,0),(1,1,0),(2,2,0)) / sqrt(3)
        float o0 = (w0*x0 - C110*(w1*x1 + w2*x2 + w3*x3)
                          + C220*(w4*x4 + w5*x5 + w6*x6 + w7*x7 + w8*x8)) * IS3;

        // l3=1 (paths (0,1,1),(1,0,1),(1,2,1),(2,1,1)) / 2
        float o1 = w0*x1 + w1*x0
                 - S*(w3*x4 + w2*x5) + T*w1*x6 + S*w1*x8
                 - S*(w4*x3 + w5*x2) + T*w6*x1 + S*w8*x1;
        float o2 = w0*x2 + w2*x0
                 - S*w1*x5 - U*w2*x6 - S*w3*x7
                 - S*w5*x1 - U*w6*x2 - S*w7*x3;
        float o3 = w0*x3 + w3*x0
                 - S*w1*x4 + T*w3*x6 - S*w2*x7 - S*w3*x8
                 - S*w4*x1 + T*w6*x3 - S*w7*x2 - S*w8*x3;
        o1 *= 0.5f; o2 *= 0.5f; o3 *= 0.5f;

        // l3=2 (paths (0,2,2),(2,0,2),(1,1,2),(2,2,2)) / 2
        float o4 = w0*x4 + w4*x0 + R*(w3*x1 + w1*x3)
                 + P*(w4*x6 + w6*x4) + Q*(w5*x7 + w7*x5);
        float o5 = w0*x5 + w5*x0 + R*(w1*x2 + w2*x1)
                 + Q*(w4*x7 + w7*x4) + RR*(w5*x6 + w6*x5) - Q*(w5*x8 + w8*x5);
        float o6 = w0*x6 + w6*x0 + G*w2*x2 - H*(w1*x1 + w3*x3)
                 + P*w4*x4 + RR*w5*x5 - P*w6*x6 + RR*w7*x7 + P*w8*x8;
        float o7 = w0*x7 + w7*x0 + R*(w3*x2 + w2*x3)
                 + Q*(w4*x5 + w5*x4) + RR*(w6*x7 + w7*x6) + Q*(w7*x8 + w8*x7);
        float o8 = w0*x8 + w8*x0 + R*(w3*x3 - w1*x1)
                 - Q*w5*x5 + Q*w7*x7 + P*(w6*x8 + w8*x6);
        o4 *= 0.5f; o5 *= 0.5f; o6 *= 0.5f; o7 *= 0.5f; o8 *= 0.5f;

        float bw = (t == 0) ? wb0 : ((t == 1) ? wb1 : wb2);
        acc[0] += o0*bw; acc[1] += o1*bw; acc[2] += o2*bw; acc[3] += o3*bw;
        acc[4] += o4*bw; acc[5] += o5*bw; acc[6] += o6*bw;
        acc[7] += o7*bw; acc[8] += o8*bw;
    }

    float* op = out + n * (DDIM * CCH) + c;   // out[n, d*128 + c]  (coalesced in c)
    #pragma unroll
    for (int d = 0; d < 9; ++d) op[d * CCH] = acc[d];
}

extern "C" void kernel_launch(void* const* d_in, const int* in_sizes, int n_in,
                              void* d_out, int out_size, void* d_ws, size_t ws_size,
                              hipStream_t stream)
{
    const float* nbody = (const float*)d_in[0];
    const float* attr  = (const float*)d_in[1];
    const float* W0    = (const float*)d_in[2];
    const float* W1    = (const float*)d_in[3];
    const float* W2    = (const float*)d_in[4];
    const float* wbody = (const float*)d_in[5];
    float* out = (float*)d_out;
    int N = in_sizes[1] / ZDIM;               // 20000

    unsigned short* wfrag = (unsigned short*)d_ws;
    float* yflat = (float*)((char*)d_ws + WFULL_OFF);

    pack_w_kernel<<<(WFRAG_ELEMS + 255) / 256, 256, 0, stream>>>(W0, W1, W2, wfrag);
    linear_wmma_kernel<<<(N + 15) / 16, 256, 0, stream>>>(attr, wfrag, yflat, N);
    cg_kernel<<<(N * CCH + 255) / 256, 256, 0, stream>>>(nbody, yflat, wbody, out, N);
}
